// PreLN_SelfAttention_seq2seq_83528523973248
// MI455X (gfx1250) — matrix operand order repaired and checked by reading, not compiled, not run on hardware
//
#include <hip/hip_runtime.h>
#include <hip/hip_bf16.h>

typedef __attribute__((ext_vector_type(16))) __bf16 v16bf;
typedef __attribute__((ext_vector_type(8)))  __bf16 v8bf;
typedef __attribute__((ext_vector_type(8)))  float  v8f;

#define S_LEN   1024
#define FEAT_N  512
#define NHEAD   8
#define DKH     64
#define NBATCH  8
#define TBLOCK  823          // S - (MAX_ITEMS+1)
#define NEGV    -999999999.0f
#define SCALE   0.125f       // 1/sqrt(64), folded into Q at store time (exact: power of 2)

static __device__ __forceinline__ v8f wmma_bf16(v16bf a, v16bf b, v8f c) {
  return __builtin_amdgcn_wmma_f32_16x16x32_bf16(false, a, false, b, (short)0, c,
                                                 false, false);
}

static __device__ __forceinline__ v16bf cat8(v8bf lo, v8bf hi) {
  return __builtin_shufflevector(lo, hi, 0, 1, 2, 3, 4, 5, 6, 7,
                                 8, 9, 10, 11, 12, 13, 14, 15);
}

// A fragment: 16x32 bf16, row-major source (row stride in elements, 16B-aligned rows).
// lanes0-15: M=lane, K {0..15}; lanes16-31: M=lane-16, K {16..31}
static __device__ __forceinline__ v16bf load_a_frag(const __bf16* src, int stride) {
  int lane = threadIdx.x & 31;
  const __bf16* row = src + (size_t)(lane & 15) * stride + ((lane >> 4) << 4);
  return cat8(*(const v8bf*)row, *(const v8bf*)(row + 8));
}

// B fragment from an N-major source: src[n][k] with row stride strideK.
// element e -> K = kb+e (kb = (lane>>4)*16), N = lane&15  => 16 contiguous elements.
static __device__ __forceinline__ v16bf load_bT_frag(const __bf16* src, int strideK) {
  int lane = threadIdx.x & 31;
  const __bf16* p = src + (size_t)(lane & 15) * strideK + ((lane >> 4) << 4);
  return cat8(*(const v8bf*)p, *(const v8bf*)(p + 8));
}

// B = K^T fragment for scores: K-dim = head-feature d, N = key index.
// kmat row-major [nkeys][DKH]; per lane: 16 contiguous features of one key row.
static __device__ __forceinline__ v16bf load_kT_frag(const __bf16* kmat, int key0, int d0) {
  int lane = threadIdx.x & 31;
  const __bf16* p = kmat + (size_t)(key0 + (lane & 15)) * DKH + d0 + ((lane >> 4) << 4);
  return cat8(*(const v8bf*)p, *(const v8bf*)(p + 8));
}

// ---- async global->LDS 16B copy (GLOBAL_LOAD_ASYNC_TO_LDS_B128, ASYNCcnt) ----
static __device__ __forceinline__ void async_copy16(const __bf16* g, __bf16* l) {
#if __has_builtin(__builtin_amdgcn_global_load_async_to_lds_b128)
  typedef int v4i __attribute__((vector_size(16)));
  typedef __attribute__((address_space(1))) v4i* gp_t;
  typedef __attribute__((address_space(3))) v4i* lp_t;
  __builtin_amdgcn_global_load_async_to_lds_b128((gp_t)g, (lp_t)l, 0, 0);
#else
  *(v8bf*)l = *(const v8bf*)g;   // fallback: VGPR round-trip staging
#endif
}

static __device__ __forceinline__ void wait_async0() {
#if __has_builtin(__builtin_amdgcn_s_wait_asynccnt)
  __builtin_amdgcn_s_wait_asynccnt(0);
#else
  asm volatile("s_wait_asynccnt 0" ::: "memory");
#endif
}

// ---------------- f32 -> bf16 conversion with [rows][cols] -> [cols][rows] transpose ----
__global__ void cvt_t_kernel(const float* __restrict__ in, __bf16* __restrict__ out,
                             int mats, int rows, int cols) {
  int idx = blockIdx.x * blockDim.x + threadIdx.x;
  int per = rows * cols;
  if (idx >= mats * per) return;
  int m = idx / per;
  int rem = idx - m * per;
  int r = rem / cols;
  int c = rem - r * cols;
  out[(size_t)m * per + (size_t)c * rows + r] = (__bf16)in[idx];
}

// ---------------- LayerNorm: one wave per token ----------------
__global__ __launch_bounds__(256) void ln_kernel(const float* __restrict__ x,
                                                 const float* __restrict__ gamma,
                                                 const float* __restrict__ beta,
                                                 __bf16* __restrict__ xn) {
  int wave = blockIdx.x * (blockDim.x >> 5) + (threadIdx.x >> 5);  // token id
  int lane = threadIdx.x & 31;
  const float* row = x + (size_t)wave * FEAT_N;
  float vals[16];
  float s = 0.f;
#pragma unroll
  for (int i = 0; i < 16; ++i) { vals[i] = row[i * 32 + lane]; s += vals[i]; }
#pragma unroll
  for (int off = 16; off >= 1; off >>= 1) s += __shfl_xor(s, off, 32);
  float mu = s * (1.0f / (float)FEAT_N);
  float vs = 0.f;
#pragma unroll
  for (int i = 0; i < 16; ++i) { float d = vals[i] - mu; vs += d * d; }
#pragma unroll
  for (int off = 16; off >= 1; off >>= 1) vs += __shfl_xor(vs, off, 32);
  float rstd = rsqrtf(vs * (1.0f / (float)FEAT_N) + 1e-5f);
  __bf16* orow = xn + (size_t)wave * FEAT_N;
#pragma unroll
  for (int i = 0; i < 16; ++i) {
    int f = i * 32 + lane;
    orow[f] = (__bf16)((vals[i] - mu) * rstd * gamma[f] + beta[f]);
  }
}

// ---------------- QKV projection: WMMA bf16 ----------------
// grid = B*H*(S/128) = 512 blocks, 8 waves; wave w -> 16-token tile, 4 DK tiles, 3 mats.
// Weights pre-transposed: wqT/wkT/wvT are [H][DK][FEAT].
// Q stored scaled by 1/sqrt(DK); V stored transposed [bh][DK][S] for the P*V stage.
__global__ __launch_bounds__(256) void qkv_kernel(const __bf16* __restrict__ xn,
                                                  const __bf16* __restrict__ wqT,
                                                  const __bf16* __restrict__ wkT,
                                                  const __bf16* __restrict__ wvT,
                                                  const float* __restrict__ bq,
                                                  const float* __restrict__ bk,
                                                  const float* __restrict__ bv,
                                                  __bf16* __restrict__ qo,
                                                  __bf16* __restrict__ ko,
                                                  __bf16* __restrict__ voT) {
  int chunk = blockIdx.x & 7;   // S/128 chunks
  int bh    = blockIdx.x >> 3;  // 0..63
  int b     = bh >> 3;
  int h     = bh & 7;
  int w     = threadIdx.x >> 5;
  int lane  = threadIdx.x & 31;
  int tok0  = chunk * 128 + w * 16;
  const __bf16* a_base = xn + ((size_t)(b * S_LEN + tok0)) * FEAT_N;
  int mrow = (lane >> 4) * 8;
  int ncol = lane & 15;

  for (int mat = 0; mat < 3; ++mat) {
    const __bf16* WT   = (mat == 0 ? wqT : mat == 1 ? wkT : wvT) + (size_t)h * DKH * FEAT_N;
    const float*  bias = (mat == 0 ? bq : mat == 1 ? bk : bv) + h * DKH;
    v8f c[4] = {};
    for (int kk = 0; kk < 16; ++kk) {
      v16bf a = load_a_frag(a_base + kk * 32, FEAT_N);
#pragma unroll
      for (int nt = 0; nt < 4; ++nt) {
        v16bf bf = load_bT_frag(WT + (size_t)(nt * 16) * FEAT_N + kk * 32, FEAT_N);
        c[nt] = wmma_bf16(a, bf, c[nt]);
      }
    }
    if (mat == 2) {
      __bf16* outT = voT + (size_t)bh * DKH * S_LEN;
#pragma unroll
      for (int nt = 0; nt < 4; ++nt)
#pragma unroll
        for (int e = 0; e < 8; ++e) {
          int col = nt * 16 + ncol;
          outT[(size_t)col * S_LEN + (tok0 + mrow + e)] = (__bf16)(c[nt][e] + bias[col]);
        }
    } else {
      __bf16* out = (mat == 0 ? qo : ko) + ((size_t)bh * S_LEN + tok0) * DKH;
      float postscale = (mat == 0) ? SCALE : 1.0f;
#pragma unroll
      for (int nt = 0; nt < 4; ++nt)
#pragma unroll
        for (int e = 0; e < 8; ++e) {
          int col = nt * 16 + ncol;
          out[(size_t)(mrow + e) * DKH + col] = (__bf16)((c[nt][e] + bias[col]) * postscale);
        }
    }
  }
}

// ---------------- Flash attention ----------------
// Block = 8 waves, all on the SAME (b,h), q-tiles 8*bx..8*bx+7. K/V tiles for each
// 32-key block are staged into LDS once per workgroup via async-to-LDS (double
// buffered, ASYNCcnt + barrier), then all waves consume them with ds_load_b128.
__global__ __launch_bounds__(256) void attn_kernel(const __bf16* __restrict__ qb,
                                                   const __bf16* __restrict__ kb,
                                                   const __bf16* __restrict__ vbT,
                                                   __bf16* __restrict__ zb) {
  __shared__ __bf16 pl[8][16 * 32];        // per-wave P staging (8 KB)
  __shared__ __bf16 kTile[2][32 * DKH];    // 2 x 4 KB: K rows [32][64]
  __shared__ __bf16 vTile[2][DKH * 32];    // 2 x 4 KB: V^T rows [64][32]

  int w    = threadIdx.x >> 5;
  int lane = threadIdx.x & 31;
  int tid  = threadIdx.x;
  int bh   = blockIdx.x >> 3;              // 8 blocks per (b,h)
  int qt   = (blockIdx.x & 7) * 8 + w;     // q-tile 0..63
  int b    = bh >> 3;
  int h    = bh & 7;
  int qbase = qt * 16;
  const __bf16* q    = qb  + (size_t)bh * S_LEN * DKH;
  const __bf16* kmat = kb  + (size_t)bh * S_LEN * DKH;
  const __bf16* vT   = vbT + (size_t)bh * DKH * S_LEN;  // [DK][S]
  int half = lane >> 4;
  int mrow = half * 8;
  int ncol = lane & 15;
  __bf16* myp = pl[w];

  // cooperative stage of one 32-key K/V tile pair (8 KB, 256 threads x 2 x 16B)
  int krow = tid >> 3, kchunk = tid & 7;   // K: 32 rows x 128B
  int vrow = tid >> 2, vchunk = tid & 3;   // V: 64 rows x 64B
#define STAGE_KV(jb, buf)                                                            \
  do {                                                                               \
    async_copy16(kmat + (size_t)((jb) + krow) * DKH + kchunk * 8,                    \
                 kTile[buf] + krow * DKH + kchunk * 8);                              \
    async_copy16(vT + (size_t)vrow * S_LEN + (jb) + vchunk * 8,                      \
                 vTile[buf] + vrow * 32 + vchunk * 8);                               \
  } while (0)

  v16bf aq0 = load_a_frag(q + (size_t)qbase * DKH + 0, DKH);
  v16bf aq1 = load_a_frag(q + (size_t)qbase * DKH + 32, DKH);

  float m_r[8], l_r[8];
#pragma unroll
  for (int e = 0; e < 8; ++e) { m_r[e] = -1e30f; l_r[e] = 0.f; }
  v8f o[4] = {};

  STAGE_KV(0, 0);
  wait_async0();
  __syncthreads();

  for (int j = 0; j < 32; ++j) {
    int jbase = j * 32;
    int buf = j & 1;
    if (j + 1 < 32) STAGE_KV(jbase + 32, buf ^ 1);   // overlap next tile load

    const __bf16* kT  = kTile[buf];
    const __bf16* vTl = vTile[buf];

    v8f s0 = {}, s1 = {};
    s0 = wmma_bf16(aq0, load_kT_frag(kT, 0, 0), s0);
    s0 = wmma_bf16(aq1, load_kT_frag(kT, 0, 32), s0);
    s1 = wmma_bf16(aq0, load_kT_frag(kT, 16, 0), s1);
    s1 = wmma_bf16(aq1, load_kT_frag(kT, 16, 32), s1);

#pragma unroll
    for (int e = 0; e < 8; ++e) {
      int i  = qbase + mrow + e;
      int j0 = jbase + ncol;
      int j1 = jbase + 16 + ncol;
      bool tail = (i >= TBLOCK) & (i <= S_LEN - 2);
      bool bl0 = ((i < TBLOCK) & (j0 >= TBLOCK)) | (tail & (j0 >= i));
      bool bl1 = ((i < TBLOCK) & (j1 >= TBLOCK)) | (tail & (j1 >= i));
      float v0 = s0[e] + (bl0 ? NEGV : 0.0f);   // softmax scale pre-folded into Q
      float v1 = s1[e] + (bl1 ? NEGV : 0.0f);

      float mx = fmaxf(v0, v1);
#pragma unroll
      for (int off = 1; off <= 8; off <<= 1) mx = fmaxf(mx, __shfl_xor(mx, off, 32));
      float mnew  = fmaxf(m_r[e], mx);
      float alpha = __expf(m_r[e] - mnew);
      float e0 = __expf(v0 - mnew);
      float e1 = __expf(v1 - mnew);
      float rs = e0 + e1;
#pragma unroll
      for (int off = 1; off <= 8; off <<= 1) rs += __shfl_xor(rs, off, 32);
      l_r[e] = l_r[e] * alpha + rs;
      m_r[e] = mnew;
#pragma unroll
      for (int nt = 0; nt < 4; ++nt) o[nt][e] *= alpha;
      int r = mrow + e;
      myp[r * 32 + ncol]      = (__bf16)e0;
      myp[r * 32 + 16 + ncol] = (__bf16)e1;
    }
    asm volatile("s_wait_dscnt 0" ::: "memory");

    // P: per-wave LDS 16x32 bf16 row-major -> A fragment (two 16B ds reads per lane)
    const __bf16* pr = myp + (lane & 15) * 32 + half * 16;
    v16bf pa = cat8(*(const v8bf*)pr, *(const v8bf*)(pr + 8));

    int kb16 = half * 16;
#pragma unroll
    for (int nt = 0; nt < 4; ++nt) {
      // B = V tile (K=key, N=dk) from LDS V^T tile: contiguous 16 keys per lane
      const __bf16* vp = vTl + (nt * 16 + ncol) * 32 + kb16;
      v16bf bvf = cat8(*(const v8bf*)vp, *(const v8bf*)(vp + 8));
      o[nt] = wmma_bf16(pa, bvf, o[nt]);
    }

    wait_async0();    // next-tile fills complete (per wave)
    __syncthreads();  // all waves done reading buf & see filled buf^1
  }

  // normalize and store z as bf16 in [B, S, H*DK] layout
  __bf16* zrow = zb + (size_t)b * S_LEN * FEAT_N + h * DKH;
#pragma unroll
  for (int e = 0; e < 8; ++e) {
    float inv = 1.0f / l_r[e];
    int tok = qbase + mrow + e;
#pragma unroll
    for (int nt = 0; nt < 4; ++nt)
      zrow[(size_t)tok * FEAT_N + nt * 16 + ncol] = (__bf16)(o[nt][e] * inv);
  }
#undef STAGE_KV
}

// ---------------- Output projection: z @ WO + bO -> f32 out ----------------
// WO pre-transposed to [FEAT_out][FEAT_in] (N-major).
__global__ __launch_bounds__(256) void proj_kernel(const __bf16* __restrict__ zb,
                                                   const __bf16* __restrict__ woT,
                                                   const float* __restrict__ bO,
                                                   float* __restrict__ out) {
  int w    = threadIdx.x >> 5;
  int lane = threadIdx.x & 31;
  int tok0 = blockIdx.x * 128 + w * 16;  // grid = 64
  const __bf16* a_base = zb + (size_t)tok0 * FEAT_N;
  int mrow = (lane >> 4) * 8;
  int ncol = lane & 15;

  for (int ng = 0; ng < 8; ++ng) {
    v8f c[4] = {};
    for (int kk = 0; kk < 16; ++kk) {
      v16bf a = load_a_frag(a_base + kk * 32, FEAT_N);
#pragma unroll
      for (int nt = 0; nt < 4; ++nt) {
        v16bf bf = load_bT_frag(woT + (size_t)((ng * 4 + nt) * 16) * FEAT_N + kk * 32,
                                FEAT_N);
        c[nt] = wmma_bf16(a, bf, c[nt]);
      }
    }
#pragma unroll
    for (int nt = 0; nt < 4; ++nt)
#pragma unroll
      for (int e = 0; e < 8; ++e) {
        int col = (ng * 4 + nt) * 16 + ncol;
        out[(size_t)(tok0 + mrow + e) * FEAT_N + col] = c[nt][e] + bO[col];
      }
  }
}

extern "C" void kernel_launch(void* const* d_in, const int* in_sizes, int n_in,
                              void* d_out, int out_size, void* d_ws, size_t ws_size,
                              hipStream_t stream) {
  const float* x  = (const float*)d_in[0];
  const float* g  = (const float*)d_in[1];
  const float* be = (const float*)d_in[2];
  const float* WQ = (const float*)d_in[3];
  const float* bq = (const float*)d_in[4];
  const float* WK = (const float*)d_in[5];
  const float* bk = (const float*)d_in[6];
  const float* WV = (const float*)d_in[7];
  const float* bv = (const float*)d_in[8];
  const float* WO = (const float*)d_in[9];
  const float* bO = (const float*)d_in[10];
  float* out = (float*)d_out;

  char* ws = (char*)d_ws;
  size_t off = 0;
  auto take = [&](size_t elems) {
    __bf16* p = (__bf16*)(ws + off);
    off += ((elems * sizeof(__bf16)) + 255) & ~(size_t)255;
    return p;
  };
  const size_t TOK = (size_t)NBATCH * S_LEN;              // 8192 tokens
  const size_t QKV = (size_t)NBATCH * NHEAD * S_LEN * DKH;
  __bf16* xn    = take(TOK * FEAT_N);
  __bf16* qbuf  = take(QKV);
  __bf16* kbuf  = take(QKV);
  __bf16* vbufT = take(QKV);                              // [bh][DK][S]
  __bf16* zbuf  = take(TOK * FEAT_N);
  __bf16* wqT   = take((size_t)NHEAD * FEAT_N * DKH);     // [H][DK][FEAT]
  __bf16* wkT   = take((size_t)NHEAD * FEAT_N * DKH);
  __bf16* wvT   = take((size_t)NHEAD * FEAT_N * DKH);
  __bf16* woT   = take((size_t)FEAT_N * FEAT_N);          // [FEAT][FEAT] transposed

  int nW = NHEAD * FEAT_N * DKH;   // 262144
  int nO = FEAT_N * FEAT_N;        // 262144
  cvt_t_kernel<<<(nW + 255) / 256, 256, 0, stream>>>(WQ, wqT, NHEAD, FEAT_N, DKH);
  cvt_t_kernel<<<(nW + 255) / 256, 256, 0, stream>>>(WK, wkT, NHEAD, FEAT_N, DKH);
  cvt_t_kernel<<<(nW + 255) / 256, 256, 0, stream>>>(WV, wvT, NHEAD, FEAT_N, DKH);
  cvt_t_kernel<<<(nO + 255) / 256, 256, 0, stream>>>(WO, woT, 1, FEAT_N, FEAT_N);

  ln_kernel<<<(int)(TOK / 8), 256, 0, stream>>>(x, g, be, xn);
  qkv_kernel<<<512, 256, 0, stream>>>(xn, wqT, wkT, wvT, bq, bk, bv, qbuf, kbuf, vbufT);
  attn_kernel<<<512, 256, 0, stream>>>(qbuf, kbuf, vbufT, zbuf);
  proj_kernel<<<64, 256, 0, stream>>>(zbuf, woT, bO, out);
}
